// PharmacoGNN_23055384445126
// MI455X (gfx1250) — matrix operand order repaired and checked by reading, not compiled, not run on hardware
//
#include <hip/hip_runtime.h>
#include <math.h>
#include <limits.h>

#define GNN_N      50000
#define GNN_E      1600000
#define GNN_DIN    128
#define GNN_DH     64
#define GNN_NCLS   10
#define GNN_NG     64
#define NEG_SLOPE  0.2f

typedef __attribute__((ext_vector_type(16))) __bf16 v16bf;
typedef __attribute__((ext_vector_type(8)))  float  v8f;

// ---------- float <-> order-preserving int (for atomicMax on floats) ----------
__device__ __forceinline__ int f2ord(float f) {
    int i = __float_as_int(f);
    return i >= 0 ? i : (i ^ 0x7FFFFFFF);
}
__device__ __forceinline__ float ord2f(int i) {
    return __int_as_float(i >= 0 ? i : (i ^ 0x7FFFFFFF));
}

// ---------- pre-pack W[K,64] (f32) into per-lane WMMA B-fragment layout (bf16) ----
// Layout: Bp[((w*kt + t)*32 + lane)*16 + i], w = 16-col tile (0..3), t = k-step,
// element = W[(t*32 + (i&7) + 8*(lane>>4) + 16*(i>>3)) * 64 + w*16 + (lane&15)]
__global__ void gnn_pack_b(const float* __restrict__ W, __bf16* __restrict__ Bp, int kt) {
    int idx = blockIdx.x * blockDim.x + threadIdx.x;
    int total = kt * 4 * 32 * 16;
    if (idx < total) {
        int i    = idx & 15;
        int lane = (idx >> 4) & 31;
        int tw   = idx >> 9;          // = w*kt + t
        int t    = tw % kt;
        int w    = tw / kt;
        int half = lane >> 4;
        int col  = (w << 4) + (lane & 15);
        int k    = t * 32 + (i & 7) + (half << 3) + ((i >> 3) << 4);
        Bp[idx] = (__bf16)W[k * GNN_DH + col];
    }
}

// ---------- WMMA GEMM: Y[M,64] = X[M,K] @ W[K,64] + bias ----------
// block = 128 threads (4 waves). wave w computes 16x16 tile at (blockIdx.x*16, w*16).
// bf16 WMMA 16x16x32, f32 accumulate. A via float4 loads, B via packed v16bf loads.
template <int K>
__global__ void gatv2_gemm_wmma(const float* __restrict__ X,
                                const __bf16* __restrict__ Bpack,
                                const float* __restrict__ bias,
                                float* __restrict__ Y) {
    constexpr int KT = K / 32;
    const int wave = threadIdx.x >> 5;
    const int lane = threadIdx.x & 31;
    const int half = lane >> 4;       // 0 or 1
    const int l16  = lane & 15;
    const int m0   = blockIdx.x * 16;
    const int n0   = wave * 16;

    const int arow = m0 + l16;        // A: both lane halves hold M=0..15, K split by half
    const float4* __restrict__ Xv =
        reinterpret_cast<const float4*>(X + (size_t)arow * K);
    const v16bf* __restrict__ Bv = reinterpret_cast<const v16bf*>(Bpack);

    v8f acc = {};
#pragma unroll
    for (int t = 0; t < KT; ++t) {
        const int q = t * 8;          // float4 index of k0 = t*32
        // A lane runs: K = k0+8*half..+7 and K = k0+16+8*half..+7 (contiguous)
        float4 r0 = Xv[q + 2 * half];
        float4 r1 = Xv[q + 2 * half + 1];
        float4 r2 = Xv[q + 4 + 2 * half];
        float4 r3 = Xv[q + 4 + 2 * half + 1];
        v16bf a;
        a[0]  = (__bf16)r0.x; a[1]  = (__bf16)r0.y; a[2]  = (__bf16)r0.z; a[3]  = (__bf16)r0.w;
        a[4]  = (__bf16)r1.x; a[5]  = (__bf16)r1.y; a[6]  = (__bf16)r1.z; a[7]  = (__bf16)r1.w;
        a[8]  = (__bf16)r2.x; a[9]  = (__bf16)r2.y; a[10] = (__bf16)r2.z; a[11] = (__bf16)r2.w;
        a[12] = (__bf16)r3.x; a[13] = (__bf16)r3.y; a[14] = (__bf16)r3.z; a[15] = (__bf16)r3.w;

        v16bf b = Bv[(wave * KT + t) * 32 + lane];

        acc = __builtin_amdgcn_wmma_f32_16x16x32_bf16(
            /*neg_a=*/false, a, /*neg_b=*/false, b,
            /*c_mod=*/(short)0, acc, /*reuse_a=*/false, /*reuse_b=*/false);
    }
    // D layout: VGPR r -> row m0 + r + 8*half, col n0 + l16
    const int col = n0 + l16;
#pragma unroll
    for (int r = 0; r < 8; ++r) {
        int row = m0 + r + 8 * half;
        Y[row * GNN_DH + col] = acc[r] + bias[col];
    }
}

// ---------- small zero ----------
__global__ void gnn_zero(float* p, int n) {
    for (int i = threadIdx.x; i < n; i += blockDim.x) p[i] = 0.f;
}

// ---------- edge_attr sum reduction ----------
__global__ void gnn_attr_reduce(const float* __restrict__ ea, float* __restrict__ out) {
    __shared__ float sh[256];
    int tid = threadIdx.x;
    float acc = 0.f;
    for (int i = blockIdx.x * blockDim.x + tid; i < GNN_E; i += gridDim.x * blockDim.x)
        acc += ea[i];
    sh[tid] = acc; __syncthreads();
    for (int s = 128; s > 0; s >>= 1) { if (tid < s) sh[tid] += sh[tid + s]; __syncthreads(); }
    if (tid == 0) atomicAdd(out, sh[0]);
}

// ---------- per-node init: mbuf=-inf(enc), sbuf=0, h = bias broadcast ----------
__global__ void gnn_init_nodes(int* __restrict__ mbuf, float* __restrict__ sbuf,
                               float* __restrict__ h, const float* __restrict__ bias) {
    int i = blockIdx.x * blockDim.x + threadIdx.x;
    if (i < GNN_N * GNN_DH) {
        h[i] = bias[i & (GNN_DH - 1)];
        if ((i & (GNN_DH - 1)) == 0) {
            int n = i >> 6;
            mbuf[n] = INT_MIN;
            sbuf[n] = 0.f;
        }
    }
}

// ---------- pass 1: logits + segment max (one wave per edge) ----------
__global__ void gnn_edge_logits(const int* __restrict__ src, const int* __restrict__ dst,
                                const float* __restrict__ eattr, const float* __restrict__ asum,
                                const float* __restrict__ We, const float* __restrict__ att,
                                const float* __restrict__ xl, const float* __restrict__ xr,
                                float* __restrict__ logits, int* __restrict__ mbuf) {
    int wid  = (blockIdx.x * blockDim.x + threadIdx.x) >> 5;
    int lane = threadIdx.x & 31;
    if (wid >= GNN_E + GNN_N) return;
    int s, d; float ea;
    if (wid < GNN_E) { s = src[wid]; d = dst[wid]; ea = eattr[wid]; }
    else             { s = d = wid - GNN_E; ea = asum[0] * (1.0f / (float)GNN_E); }

    const float2 vl = *(const float2*)(xl + s * GNN_DH + lane * 2);
    const float2 vr = *(const float2*)(xr + d * GNN_DH + lane * 2);
    const float2 we = *(const float2*)(We + lane * 2);
    const float2 at = *(const float2*)(att + lane * 2);

    float v0 = vl.x + vr.x + ea * we.x;
    float v1 = vl.y + vr.y + ea * we.y;
    v0 = v0 > 0.f ? v0 : NEG_SLOPE * v0;          // leaky_relu
    v1 = v1 > 0.f ? v1 : NEG_SLOPE * v1;
    float part = v0 * at.x + v1 * at.y;
#pragma unroll
    for (int off = 16; off > 0; off >>= 1) part += __shfl_xor(part, off, 32);
    if (lane == 0) {
        logits[wid] = part;
        atomicMax(&mbuf[d], f2ord(part));
    }
}

// ---------- pass 2: z = exp(logit - m[d]); segment sum ----------
__global__ void gnn_edge_expsum(const int* __restrict__ dst, const float* __restrict__ logits,
                                const int* __restrict__ mbuf, float* __restrict__ sbuf) {
    int e = blockIdx.x * blockDim.x + threadIdx.x;
    if (e >= GNN_E + GNN_N) return;
    int d = (e < GNN_E) ? dst[e] : (e - GNN_E);
    float z = __expf(logits[e] - ord2f(mbuf[d]));
    atomicAdd(&sbuf[d], z);
}

// ---------- pass 3: out[d] += alpha * xl[s] (one wave per edge) ----------
__global__ void gnn_edge_aggregate(const int* __restrict__ src, const int* __restrict__ dst,
                                   const float* __restrict__ logits, const int* __restrict__ mbuf,
                                   const float* __restrict__ sbuf, const float* __restrict__ xl,
                                   float* __restrict__ out) {
    int wid  = (blockIdx.x * blockDim.x + threadIdx.x) >> 5;
    int lane = threadIdx.x & 31;
    if (wid >= GNN_E + GNN_N) return;
    int s, d;
    if (wid < GNN_E) { s = src[wid]; d = dst[wid]; }
    else             { s = d = wid - GNN_E; }
    float alpha = __expf(logits[wid] - ord2f(mbuf[d])) / (sbuf[d] + 1e-16f);
    const float2 vl = *(const float2*)(xl + s * GNN_DH + lane * 2);
    atomicAdd(&out[d * GNN_DH + lane * 2 + 0], alpha * vl.x);
    atomicAdd(&out[d * GNN_DH + lane * 2 + 1], alpha * vl.y);
}

// ---------- ELU in place ----------
__global__ void gnn_elu(float* __restrict__ h, int n) {
    int i = blockIdx.x * blockDim.x + threadIdx.x;
    if (i < n) {
        float v = h[i];
        h[i] = v > 0.f ? v : (__expf(v) - 1.f);
    }
}

// ---------- global mean pool accumulation ----------
__global__ void gnn_pool(const float* __restrict__ h, const int* __restrict__ batch,
                         float* __restrict__ gsum, float* __restrict__ gcnt) {
    int i = blockIdx.x * blockDim.x + threadIdx.x;
    if (i < GNN_N * GNN_DH) {
        int n = i >> 6, d = i & (GNN_DH - 1);
        int g = batch[n];
        atomicAdd(&gsum[g * GNN_DH + d], h[i]);
        if (d == 0) atomicAdd(&gcnt[g], 1.f);
    }
}

// ---------- classifier: out[G,10] = (gsum/cnt) @ Wc + bc ----------
__global__ void gnn_classify(const float* __restrict__ gsum, const float* __restrict__ gcnt,
                             const float* __restrict__ Wc, const float* __restrict__ bc,
                             float* __restrict__ out) {
    int t = threadIdx.x;
    if (t < GNN_NG * GNN_NCLS) {
        int g = t / GNN_NCLS, c = t % GNN_NCLS;
        float inv = 1.0f / fmaxf(gcnt[g], 1.f);
        float acc = bc[c];
        for (int d = 0; d < GNN_DH; ++d)
            acc += (gsum[g * GNN_DH + d] * inv) * Wc[d * GNN_NCLS + c];
        out[t] = acc;
    }
}

extern "C" void kernel_launch(void* const* d_in, const int* in_sizes, int n_in,
                              void* d_out, int out_size, void* d_ws, size_t ws_size,
                              hipStream_t stream) {
    const float* x     = (const float*)d_in[0];
    const int*   eidx  = (const int*)d_in[1];
    const float* eattr = (const float*)d_in[2];
    const int*   batch = (const int*)d_in[3];
    const float* Wl1 = (const float*)d_in[4];  const float* bl1 = (const float*)d_in[5];
    const float* Wr1 = (const float*)d_in[6];  const float* br1 = (const float*)d_in[7];
    const float* We1 = (const float*)d_in[8];  const float* att1 = (const float*)d_in[9];
    const float* b1  = (const float*)d_in[10];
    const float* Wl2 = (const float*)d_in[11]; const float* bl2 = (const float*)d_in[12];
    const float* Wr2 = (const float*)d_in[13]; const float* br2 = (const float*)d_in[14];
    const float* We2 = (const float*)d_in[15]; const float* att2 = (const float*)d_in[16];
    const float* b2  = (const float*)d_in[17];
    const float* Wc  = (const float*)d_in[18]; const float* bc  = (const float*)d_in[19];
    float* out = (float*)d_out;

    const int* src = eidx;
    const int* dst = eidx + GNN_E;

    // workspace layout (floats)
    float* ws = (float*)d_ws;
    float* xl     = ws;                               // N*64
    float* xr     = xl + (size_t)GNN_N * GNN_DH;      // N*64
    float* h      = xr + (size_t)GNN_N * GNN_DH;      // N*64
    float* logits = h  + (size_t)GNN_N * GNN_DH;      // E+N
    int*   mbuf   = (int*)(logits + (GNN_E + GNN_N)); // N (int)
    float* sbuf   = (float*)(mbuf + GNN_N);           // N
    float* gsum   = sbuf + GNN_N;                     // 64*64
    float* gcnt   = gsum + GNN_NG * GNN_DH;           // 64
    float* asum   = gcnt + GNN_NG;                    // 1
    size_t bpOff  = (size_t)(asum + 1 - ws);
    bpOff = (bpOff + 7) & ~(size_t)7;                 // 32-byte align for v16bf loads
    __bf16* bpack = (__bf16*)(ws + bpOff);            // up to 4*128*16 bf16 = 16 KB

    const int tilesM    = GNN_N / 16;                       // 3125
    const int nodeElems = GNN_N * GNN_DH;                   // 3.2M
    const int gridNode  = (nodeElems + 255) / 256;
    const int totalEdge = GNN_E + GNN_N;
    const int gridWave  = (totalEdge + 7) / 8;              // 8 waves per 256-thread block
    const int gridEdge  = (totalEdge + 255) / 256;
    const int packElems128 = 4 * 4 * 32 * 16;               // kt=4
    const int packElems64  = 4 * 2 * 32 * 16;               // kt=2

    // zero pool accumulators + attr sum (contiguous: gsum, gcnt, asum)
    gnn_zero<<<1, 256, 0, stream>>>(gsum, GNN_NG * GNN_DH + GNN_NG + 1);
    gnn_attr_reduce<<<512, 256, 0, stream>>>(eattr, asum);

    // ---- layer 1 ----
    gnn_pack_b<<<(packElems128 + 255) / 256, 256, 0, stream>>>(Wl1, bpack, 4);
    gatv2_gemm_wmma<GNN_DIN><<<tilesM, 128, 0, stream>>>(x, bpack, bl1, xl);
    gnn_pack_b<<<(packElems128 + 255) / 256, 256, 0, stream>>>(Wr1, bpack, 4);
    gatv2_gemm_wmma<GNN_DIN><<<tilesM, 128, 0, stream>>>(x, bpack, br1, xr);
    gnn_init_nodes<<<gridNode, 256, 0, stream>>>(mbuf, sbuf, h, b1);
    gnn_edge_logits<<<gridWave, 256, 0, stream>>>(src, dst, eattr, asum, We1, att1, xl, xr, logits, mbuf);
    gnn_edge_expsum<<<gridEdge, 256, 0, stream>>>(dst, logits, mbuf, sbuf);
    gnn_edge_aggregate<<<gridWave, 256, 0, stream>>>(src, dst, logits, mbuf, sbuf, xl, h);
    gnn_elu<<<gridNode, 256, 0, stream>>>(h, nodeElems);

    // ---- layer 2 (h is input; xl/xr reused) ----
    gnn_pack_b<<<(packElems64 + 255) / 256, 256, 0, stream>>>(Wl2, bpack, 2);
    gatv2_gemm_wmma<GNN_DH><<<tilesM, 128, 0, stream>>>(h, bpack, bl2, xl);
    gnn_pack_b<<<(packElems64 + 255) / 256, 256, 0, stream>>>(Wr2, bpack, 2);
    gatv2_gemm_wmma<GNN_DH><<<tilesM, 128, 0, stream>>>(h, bpack, br2, xr);
    gnn_init_nodes<<<gridNode, 256, 0, stream>>>(mbuf, sbuf, h, b2);   // h1 dead after GEMMs
    gnn_edge_logits<<<gridWave, 256, 0, stream>>>(src, dst, eattr, asum, We2, att2, xl, xr, logits, mbuf);
    gnn_edge_expsum<<<gridEdge, 256, 0, stream>>>(dst, logits, mbuf, sbuf);
    gnn_edge_aggregate<<<gridWave, 256, 0, stream>>>(src, dst, logits, mbuf, sbuf, xl, h);
    gnn_elu<<<gridNode, 256, 0, stream>>>(h, nodeElems);

    // ---- pool + classify ----
    gnn_pool<<<gridNode, 256, 0, stream>>>(h, batch, gsum, gcnt);
    gnn_classify<<<1, 640, 0, stream>>>(gsum, gcnt, Wc, bc, out);
}
